// Model_3143916061187
// MI455X (gfx1250) — matrix operand order repaired and checked by reading, not compiled, not run on hardware
//
#include <hip/hip_runtime.h>
#include <hip/hip_bf16.h>

typedef __bf16 bf16_t;
typedef __attribute__((ext_vector_type(16))) __bf16 v16bf;
typedef __attribute__((ext_vector_type(8)))  float  v8f;
typedef __attribute__((ext_vector_type(4)))  float  vfloat4;

union Frag16 { v16bf v; vfloat4 q[2]; };

// ---------- ordered-uint encoding for float atomicMax ----------
__device__ __forceinline__ unsigned f2ord(float f) {
    unsigned u = __float_as_uint(f);
    return (u & 0x80000000u) ? ~u : (u | 0x80000000u);
}
__device__ __forceinline__ float ord2f(unsigned u) {
    return __uint_as_float((u & 0x80000000u) ? (u & 0x7FFFFFFFu) : ~u);
}

// ---------- elementwise prep kernels ----------
__global__ void make_xa_bf16(const float* __restrict__ x, const float* __restrict__ n1,
                             const float* __restrict__ n2, bf16_t* __restrict__ xa,
                             long long n) {
    long long i = (long long)blockIdx.x * blockDim.x + threadIdx.x;
    if (i < n) xa[i] = (bf16_t)(x[i] + n1[i] + n2[i]);
}

__global__ void f32_to_bf16(const float* __restrict__ in, bf16_t* __restrict__ out, long long n) {
    long long i = (long long)blockIdx.x * blockDim.x + threadIdx.x;
    if (i < n) out[i] = (bf16_t)in[i];
}

// W: [K x Nc] row-major  ->  Wt: [Nc x K] row-major, bf16
__global__ void transpose_to_bf16(const float* __restrict__ W, bf16_t* __restrict__ Wt,
                                  int K, int Nc) {
    int idx = blockIdx.x * blockDim.x + threadIdx.x;
    if (idx >= K * Nc) return;
    int c = idx / K, k = idx - c * K;
    Wt[idx] = (bf16_t)W[k * Nc + c];
}

// ---------- degree / norm ----------
__global__ void deg_init(float* __restrict__ deg, int n) {
    int i = blockIdx.x * blockDim.x + threadIdx.x;
    if (i < n) deg[i] = 1.0f;                 // self-loop contribution
}
__global__ void deg_scatter(const long long* __restrict__ dst, float* __restrict__ deg, long long E) {
    long long e = (long long)blockIdx.x * blockDim.x + threadIdx.x;
    if (e < E) atomicAdd(&deg[dst[e]], 1.0f);
}
__global__ void deg_to_dinv(float* __restrict__ deg, int n) {
    int i = blockIdx.x * blockDim.x + threadIdx.x;
    if (i < n) deg[i] = rsqrtf(deg[i]);       // deg >= 1 always (self-loop)
}
__global__ void edge_norm(const long long* __restrict__ src, const long long* __restrict__ dst,
                          const float* __restrict__ dinv, float* __restrict__ nrm, long long E) {
    long long e = (long long)blockIdx.x * blockDim.x + threadIdx.x;
    if (e < E) nrm[e] = dinv[src[e]] * dinv[dst[e]];
}

// ---------- WMMA GEMM:  C[M x Nc] = A[M x K] * B,  Bt = B^T as [Nc x K] bf16 ----------
__global__ void gemm_bf16_wmma(const bf16_t* __restrict__ A, const bf16_t* __restrict__ Bt,
                               float* __restrict__ C, int M, int K, int Nc) {
    int wave = (blockIdx.x * blockDim.x + threadIdx.x) >> 5;
    int lane = threadIdx.x & 31;
    int ntiles = Nc >> 4, mtiles = M >> 4;
    if (wave >= mtiles * ntiles) return;
    int tm = wave / ntiles;
    int tn = wave - tm * ntiles;
    int row = (tm << 4) + (lane & 15);
    int col = (tn << 4) + (lane & 15);
    int koffA = (lane < 16) ? 0 : 8;
    int koffB = (lane < 16) ? 0 : 8;
    const bf16_t* arow = A + (size_t)row * K;
    const bf16_t* brow = Bt + (size_t)col * K;
    v8f acc = {};
    for (int kk = 0; kk < K; kk += 32) {
        Frag16 a, b;
        a.q[0] = *(const vfloat4*)(arow + kk + koffA);
        a.q[1] = *(const vfloat4*)(arow + kk + 16 + koffA);
        b.q[0] = *(const vfloat4*)(brow + kk + koffB);
        b.q[1] = *(const vfloat4*)(brow + kk + koffB + 16);
        acc = __builtin_amdgcn_wmma_f32_16x16x32_bf16(false, a.v, false, b.v,
                                                      (short)0, acc, false, false);
    }
    float* crow = C + (size_t)((tm << 4) + ((lane < 16) ? 0 : 8)) * Nc + col;
#pragma unroll
    for (int r = 0; r < 8; r++) crow[(size_t)r * Nc] = acc[r];
}

// ---------- scatter-add stage: out = bias + self-loop + sum_edges ----------
__global__ void node_init(const float* __restrict__ h, const float* __restrict__ dinv,
                          const float* __restrict__ bias, float* __restrict__ out,
                          long long n, int D) {
    long long idx = (long long)blockIdx.x * blockDim.x + threadIdx.x;
    if (idx >= n * D) return;
    long long i = idx / D;
    int c = (int)(idx - i * D);
    float di = dinv[i];
    out[idx] = bias[c] + h[idx] * di * di;
}

template <int D4>   // D/4 channel-quads
__global__ void edge_scatter(const long long* __restrict__ src, const long long* __restrict__ dst,
                             const float* __restrict__ nrm, const float* __restrict__ h,
                             float* __restrict__ out, long long E) {
    long long idx = (long long)blockIdx.x * blockDim.x + threadIdx.x;
    if (idx >= E * D4) return;
    long long e = idx / D4;
    int q = (int)(idx - e * D4);
    long long s = src[e], d = dst[e];
    float w = nrm[e];
    vfloat4 v = *(const vfloat4*)(h + s * (D4 * 4) + q * 4);
    float* o = out + d * (D4 * 4) + q * 4;
    atomicAdd(o + 0, v.x * w);
    atomicAdd(o + 1, v.y * w);
    atomicAdd(o + 2, v.z * w);
    atomicAdd(o + 3, v.w * w);
}

// ---------- fused concat-GEMM + global max-pool (never materializes h_points) ----------
__global__ void pool_init(unsigned* __restrict__ pool) {
    pool[threadIdx.x] = f2ord(-3.402823466e38f);
}

__global__ void gemm_concat_maxpool(const bf16_t* __restrict__ X1,   // [M x 64]
                                    const bf16_t* __restrict__ X2,   // [M x 128]
                                    const bf16_t* __restrict__ Wlt,  // [128 x 192] (Wl^T)
                                    const float* __restrict__ bl,
                                    unsigned* __restrict__ pool, int M) {
    int wave = (blockIdx.x * blockDim.x + threadIdx.x) >> 5;
    int lane = threadIdx.x & 31;
    const int ntiles = 8;                       // 128 / 16
    int mtiles = M >> 4;
    if (wave >= mtiles * ntiles) return;
    int tm = wave / ntiles;
    int tn = wave - tm * ntiles;
    int row = (tm << 4) + (lane & 15);
    int col = (tn << 4) + (lane & 15);
    int koffA = (lane < 16) ? 0 : 8;
    int koffB = (lane < 16) ? 0 : 8;
    const bf16_t* a1 = X1 + (size_t)row * 64;
    const bf16_t* a2 = X2 + (size_t)row * 128;
    const bf16_t* brow = Wlt + (size_t)col * 192;
    v8f acc = {};
#pragma unroll
    for (int kk = 0; kk < 64; kk += 32) {       // x1 part: Wl rows [0,64)
        Frag16 a, b;
        a.q[0] = *(const vfloat4*)(a1 + kk + koffA);
        a.q[1] = *(const vfloat4*)(a1 + kk + 16 + koffA);
        b.q[0] = *(const vfloat4*)(brow + kk + koffB);
        b.q[1] = *(const vfloat4*)(brow + kk + koffB + 16);
        acc = __builtin_amdgcn_wmma_f32_16x16x32_bf16(false, a.v, false, b.v,
                                                      (short)0, acc, false, false);
    }
#pragma unroll
    for (int kk = 0; kk < 128; kk += 32) {      // x2 part: Wl rows [64,192)
        Frag16 a, b;
        a.q[0] = *(const vfloat4*)(a2 + kk + koffA);
        a.q[1] = *(const vfloat4*)(a2 + kk + 16 + koffA);
        b.q[0] = *(const vfloat4*)(brow + 64 + kk + koffB);
        b.q[1] = *(const vfloat4*)(brow + 64 + kk + koffB + 16);
        acc = __builtin_amdgcn_wmma_f32_16x16x32_bf16(false, a.v, false, b.v,
                                                      (short)0, acc, false, false);
    }
    float bb = bl[col];
    float m = -3.402823466e38f;
#pragma unroll
    for (int r = 0; r < 8; r++) m = fmaxf(m, acc[r] + bb);
    atomicMax(pool + col, f2ord(m));            // 1 atomic per lane, column-keyed
}

// ---------- tiny final MLP on the pooled 128-vector; writes all 4 outputs ----------
__global__ void mlp_finalize(const unsigned* __restrict__ pool,
                             const float* __restrict__ W3, const float* __restrict__ b3,
                             const float* __restrict__ W4, const float* __restrict__ b4,
                             float* __restrict__ out) {
    __shared__ float h[128];
    __shared__ float hid[256];
    int t = threadIdx.x;
    if (t < 128) h[t] = ord2f(pool[t]);
    __syncthreads();
    float s = b3[t];
    for (int i = 0; i < 128; i++) s += h[i] * W3[i * 256 + t];
    hid[t] = fmaxf(s, 0.0f);
    __syncthreads();
    if (t < 128) {
        float o = b4[t];
        for (int i = 0; i < 256; i++) o += hid[i] * W4[i * 128 + t];
        out[t]       = h[t];   // h_1
        out[128 + t] = h[t];   // h_2  (branches are identical)
        out[256 + t] = o;      // compact_h_1
        out[384 + t] = o;      // compact_h_2
    }
}

extern "C" void kernel_launch(void* const* d_in, const int* in_sizes, int n_in,
                              void* d_out, int out_size, void* d_ws, size_t ws_size,
                              hipStream_t stream) {
    const float*     x   = (const float*)d_in[0];
    const long long* ei  = (const long long*)d_in[1];   // int64 per reference
    const float*     nz1 = (const float*)d_in[2];
    const float*     nz2 = (const float*)d_in[3];
    const float* W1 = (const float*)d_in[4];  const float* b1 = (const float*)d_in[5];
    const float* W2 = (const float*)d_in[6];  const float* b2 = (const float*)d_in[7];
    const float* Wl = (const float*)d_in[8];  const float* bl = (const float*)d_in[9];
    const float* W3 = (const float*)d_in[10]; const float* b3 = (const float*)d_in[11];
    const float* W4 = (const float*)d_in[12]; const float* b4 = (const float*)d_in[13];

    const long long N = in_sizes[0] / 64;
    const long long E = in_sizes[1] / 2;
    const long long* srcIdx = ei;
    const long long* dstIdx = ei + E;

    // workspace arena (256B aligned slices)
    char* ws = (char*)d_ws;
    size_t off = 0;
    auto arena = [&](size_t bytes) -> void* {
        void* p = ws + off;
        off = (off + bytes + 255) & ~(size_t)255;
        return p;
    };
    bf16_t*   xa_bf = (bf16_t*)  arena((size_t)N * 64 * 2);
    float*    dinv  = (float*)   arena((size_t)N * 4);
    float*    nrm   = (float*)   arena((size_t)E * 4);
    float*    h1    = (float*)   arena((size_t)N * 64 * 4);
    float*    x1f   = (float*)   arena((size_t)N * 64 * 4);
    bf16_t*   x1b   = (bf16_t*)  arena((size_t)N * 64 * 2);
    float*    h2    = (float*)   arena((size_t)N * 128 * 4);
    float*    x2f   = (float*)   arena((size_t)N * 128 * 4);
    bf16_t*   x2b   = (bf16_t*)  arena((size_t)N * 128 * 2);
    bf16_t*   w1t   = (bf16_t*)  arena(64 * 64 * 2);
    bf16_t*   w2t   = (bf16_t*)  arena(128 * 64 * 2);
    bf16_t*   wlt   = (bf16_t*)  arena(128 * 192 * 2);
    unsigned* pool  = (unsigned*)arena(128 * 4);

    const int B = 256;
    auto blocks = [&](long long total) { return (unsigned)((total + B - 1) / B); };

    // prep: xa (f32 sum -> bf16), transposed bf16 weights, degree/norm
    make_xa_bf16<<<blocks(N * 64), B, 0, stream>>>(x, nz1, nz2, xa_bf, N * 64);
    transpose_to_bf16<<<blocks(64 * 64),   B, 0, stream>>>(W1, w1t, 64, 64);
    transpose_to_bf16<<<blocks(64 * 128),  B, 0, stream>>>(W2, w2t, 64, 128);
    transpose_to_bf16<<<blocks(192 * 128), B, 0, stream>>>(Wl, wlt, 192, 128);
    deg_init<<<blocks(N), B, 0, stream>>>(dinv, (int)N);
    deg_scatter<<<blocks(E), B, 0, stream>>>(dstIdx, dinv, E);
    deg_to_dinv<<<blocks(N), B, 0, stream>>>(dinv, (int)N);
    edge_norm<<<blocks(E), B, 0, stream>>>(srcIdx, dstIdx, dinv, nrm, E);

    // conv1: h1 = xa @ W1 (WMMA), then normalized scatter-add + bias
    gemm_bf16_wmma<<<blocks((N / 16) * 4 * 32), B, 0, stream>>>(xa_bf, w1t, h1, (int)N, 64, 64);
    node_init<<<blocks(N * 64), B, 0, stream>>>(h1, dinv, b1, x1f, N, 64);
    edge_scatter<16><<<blocks(E * 16), B, 0, stream>>>(srcIdx, dstIdx, nrm, h1, x1f, E);
    f32_to_bf16<<<blocks(N * 64), B, 0, stream>>>(x1f, x1b, N * 64);

    // conv2: h2 = x1 @ W2 (WMMA), scatter-add + bias
    gemm_bf16_wmma<<<blocks((N / 16) * 8 * 32), B, 0, stream>>>(x1b, w2t, h2, (int)N, 64, 128);
    node_init<<<blocks(N * 128), B, 0, stream>>>(h2, dinv, b2, x2f, N, 128);
    edge_scatter<32><<<blocks(E * 32), B, 0, stream>>>(srcIdx, dstIdx, nrm, h2, x2f, E);
    f32_to_bf16<<<blocks(N * 128), B, 0, stream>>>(x2f, x2b, N * 128);

    // fused concat-GEMM + global max-pool, then tiny MLP head
    pool_init<<<1, 128, 0, stream>>>(pool);
    gemm_concat_maxpool<<<blocks((N / 16) * 8 * 32), B, 0, stream>>>(x1b, x2b, wlt, bl, pool, (int)N);
    mlp_finalize<<<1, 256, 0, stream>>>(pool, W3, b3, W4, b4, (float*)d_out);
}
